// HadamardW4A4LinearPyTorch_1176821039708
// MI455X (gfx1250) — 8-bit matrix operand order repaired and checked by reading, compiled, not run on hardware
//
#include <hip/hip_runtime.h>
#include <stdint.h>

#define IN_FEAT  4096
#define OUT_FEAT 4096

typedef int v8i __attribute__((ext_vector_type(8)));

__device__ __forceinline__ uint32_t pack4(int b0, int b1, int b2, int b3) {
    return (uint32_t)(b0 & 0xff) | ((uint32_t)(b1 & 0xff) << 8) |
           ((uint32_t)(b2 & 0xff) << 16) | ((uint32_t)(b3 & 0xff) << 24);
}

__global__ __launch_bounds__(256) void quantize_swizzle_x(
    const float* __restrict__ x, uint32_t* __restrict__ xq,
    float* __restrict__ xscale)
{
    __shared__ float red[256];
    const int t   = blockIdx.x;
    const int tid = threadIdx.x;
    const float* row = x + (size_t)t * IN_FEAT;

    float4 f[4];
    const float4* rp = (const float4*)(row + tid * 16);
    f[0] = rp[0]; f[1] = rp[1]; f[2] = rp[2]; f[3] = rp[3];

    float m = 0.0f;
#pragma unroll
    for (int i = 0; i < 4; ++i) {
        m = fmaxf(m, fabsf(f[i].x)); m = fmaxf(m, fabsf(f[i].y));
        m = fmaxf(m, fabsf(f[i].z)); m = fmaxf(m, fabsf(f[i].w));
    }
    red[tid] = m;
    __syncthreads();
#pragma unroll
    for (int s = 128; s > 0; s >>= 1) {
        if (tid < s) red[tid] = fmaxf(red[tid], red[tid + s]);
        __syncthreads();
    }
    const float scale = fmaxf(red[0] * (1.0f / 7.0f), 1e-5f);
    if (tid == 0) xscale[t] = scale;
    const float inv = 1.0f / scale;

    int q[16];
#pragma unroll
    for (int i = 0; i < 4; ++i) {
        const float* e = (const float*)&f[i];
#pragma unroll
        for (int j = 0; j < 4; ++j) {
            int v = (int)rintf(e[j] * inv);
            v = v < -8 ? -8 : (v > 7 ? 7 : v);
            q[i * 4 + j] = v;
        }
    }

    const int mt = t >> 4, mrow = t & 15;
    const int kt = tid >> 2, q16 = tid & 3;
    uint32_t* tp = xq + ((size_t)mt * 64 + kt) * 256;
    tp[(size_t)mrow * 8 + 2 * q16]            = pack4(q[0],  q[1],  q[2],  q[3]);
    tp[(size_t)mrow * 8 + 2 * q16 + 1]        = pack4(q[4],  q[5],  q[6],  q[7]);
    tp[(size_t)(mrow + 16) * 8 + 2 * q16]     = pack4(q[8],  q[9],  q[10], q[11]);
    tp[(size_t)(mrow + 16) * 8 + 2 * q16 + 1] = pack4(q[12], q[13], q[14], q[15]);
}

__global__ __launch_bounds__(256) void unpack_swizzle_w(
    const int* __restrict__ wp, uint32_t* __restrict__ wq)
{
    const uint32_t gid = blockIdx.x * 256u + threadIdx.x;
    const int v    = gid & 7;
    const int lane = (gid >> 3) & 31;
    const int kt   = (gid >> 8) & 63;
    const int nt   = gid >> 14;

    const int n  = nt * 16 + (lane & 15);
    const int kh = lane >> 4;
    const int k  = kt * 64 + (v >> 1) * 16 + kh * 8 + (v & 1) * 4;

    const int* p = wp + (size_t)n * (IN_FEAT / 2) + (k >> 1);
    const int p0 = p[0], p1 = p[1];
    const int b0 = ((p0        & 15) ^ 8) - 8;
    const int b1 = (((p0 >> 4) & 15) ^ 8) - 8;
    const int b2 = ((p1        & 15) ^ 8) - 8;
    const int b3 = (((p1 >> 4) & 15) ^ 8) - 8;
    wq[gid] = pack4(b0, b1, b2, b3);
}

__global__ __launch_bounds__(256) void gemm_w4a4_wmma(
    const uint32_t* __restrict__ xq, const uint32_t* __restrict__ wq,
    const float* __restrict__ xscale, const float* __restrict__ wscale,
    float* __restrict__ out)
{
    const int tid  = threadIdx.x;
    const int lane = tid & 31;
    const int wave = tid >> 5;
    const int wm   = wave & 1;
    const int wn   = wave >> 1;
    const int bn   = blockIdx.x;
    const int bm   = blockIdx.y;

    const int mt0 = bm * 8 + wm * 4;
    const int nt0 = bn * 16 + wn * 4;

    const uint32_t* aBase = xq + (size_t)mt0 * 64 * 256 + lane * 8;
    const uint32_t* bBase = wq + (size_t)nt0 * 64 * 256 + lane * 8;

    v8i acc[4][4];
#pragma unroll
    for (int i = 0; i < 4; ++i)
#pragma unroll
        for (int j = 0; j < 4; ++j)
            acc[i][j] = (v8i)0;

    for (int kt = 0; kt < 64; ++kt) {
        __builtin_prefetch((const void*)(aBase + ((size_t)kt + 8) * 256), 0, 3);
        __builtin_prefetch((const void*)(bBase + ((size_t)kt + 8) * 256), 0, 3);

        v8i a[4], b[4];
#pragma unroll
        for (int i = 0; i < 4; ++i)
            a[i] = *(const v8i*)(aBase + ((size_t)i * 64 + kt) * 256);
#pragma unroll
        for (int j = 0; j < 4; ++j)
            b[j] = *(const v8i*)(bBase + ((size_t)j * 64 + kt) * 256);

#pragma unroll
        for (int i = 0; i < 4; ++i)
#pragma unroll
            for (int j = 0; j < 4; ++j)
                acc[i][j] = __builtin_amdgcn_wmma_i32_16x16x64_iu8(
                     true, a[i],  true, b[j],
                    acc[i][j],  false,  false);
    }

    const int nlo = lane & 15;
    const int hi  = lane >> 4;
#pragma unroll
    for (int j = 0; j < 4; ++j) {
        const int ng = (nt0 + j) * 16 + nlo;
        const float wsc = wscale[ng];
#pragma unroll
        for (int i = 0; i < 4; ++i) {
            const int tbase = (mt0 + i) * 16 + 8 * hi;
#pragma unroll
            for (int r = 0; r < 8; ++r) {
                const int tg = tbase + r;
                out[(size_t)tg * OUT_FEAT + ng] =
                    (float)acc[i][j][r] * xscale[tg] * wsc;
            }
        }
    }
}

extern "C" void kernel_launch(void* const* d_in, const int* in_sizes, int n_in,
                              void* d_out, int out_size, void* d_ws, size_t ws_size,
                              hipStream_t stream) {
    (void)n_in; (void)out_size; (void)ws_size;
    const float* x   = (const float*)d_in[0];
    const int*   wp  = (const int*)d_in[1];
    const float* wsc = (const float*)d_in[2];
    float*       out = (float*)d_out;

    const int T = in_sizes[0] / IN_FEAT;

    const size_t wq_bytes = (size_t)OUT_FEAT * IN_FEAT;
    const size_t xq_bytes = (size_t)T * IN_FEAT;
    uint32_t* wq     = (uint32_t*)d_ws;
    uint32_t* xq     = (uint32_t*)((char*)d_ws + wq_bytes);
    float*    xscale = (float*)((char*)d_ws + wq_bytes + xq_bytes);

    quantize_swizzle_x<<<T, 256, 0, stream>>>(x, xq, xscale);
    unpack_swizzle_w<<<(OUT_FEAT * (IN_FEAT / 4)) / 256, 256, 0, stream>>>(wp, wq);
    gemm_w4a4_wmma<<<dim3(OUT_FEAT / 256, T / 128), 256, 0, stream>>>(
        xq, wq, xscale, wsc, out);
}
